// Multiheaded_Attention_10153302688113
// MI455X (gfx1250) — compile-verified, matrix operand order checked by reading, not run on hardware
//
#include <hip/hip_runtime.h>
#include <hip/hip_bf16.h>

// ---------------- problem constants (from reference) ----------------
#define NB_    8
#define QS_    512
#define VS_    512
#define H_     8
#define D_     64
#define MD_    512          // model dim = H_*D_
#define LMAX_  1024
#define RELROWS_ (2*LMAX_+1)
#define SCALE_ 0.125f       // 64^-0.5

// ---------------- vector types for WMMA ----------------
typedef __attribute__((ext_vector_type(16))) __bf16         v16bf;
typedef __attribute__((ext_vector_type(8)))  float          v8f;
typedef __attribute__((ext_vector_type(8)))  unsigned short v8us;
typedef __attribute__((ext_vector_type(16))) unsigned short v16us;

// Pack two f32 into two bf16 (round-to-nearest, ties-up) in ONE v_perm_b32:
// round both 32-bit values, then extract the two high halves.
__device__ __forceinline__ unsigned int pack2bf(float lo, float hi) {
  unsigned int u0 = __builtin_bit_cast(unsigned int, lo) + 0x8000u;
  unsigned int u1 = __builtin_bit_cast(unsigned int, hi) + 0x8000u;
  // result = { u1[31:16], u0[31:16] }
  return __builtin_amdgcn_perm(u1, u0, 0x07060302u);
}

// Build one 16-element bf16 fragment from two 16B-aligned LDS runs.
// CDNA5 16-bit A/B layout: lane (0..15 / 16..31) holds K in two 8-element
// runs: [kh*8, kh*8+8) and [16+kh*8, 16+kh*8+8)  (kh = lane>>4).
__device__ __forceinline__ v16bf frag16(const unsigned int* p0,
                                        const unsigned int* p1) {
  v8us a = *(const v8us*)p0;
  v8us b = *(const v8us*)p1;
  v16us c = __builtin_shufflevector(a, b, 0,1,2,3,4,5,6,7,
                                          8,9,10,11,12,13,14,15);
  return __builtin_bit_cast(v16bf, c);
}

// =====================================================================
// Generic C = A @ B (row-major, f32 in/out, bf16 WMMA, double-buffered LDS).
// 64x64 C tile per block, 256 threads = 8 waves, each wave 16x32 of C.
// Batched via blockIdx.z: off = (z/zmod)*s1 + (z%zmod)*s2.
// Requires N % 64 == 0, K % 32 == 0; M is guarded.
// =====================================================================
__global__ __launch_bounds__(256) void gemm_nn(
    const float* __restrict__ Ab, const float* __restrict__ Bb,
    float* __restrict__ Cb,
    int M, int N, int K, int lda, int ldb, int ldc,
    int zmod, long long sA1, long long sA2,
    long long sB1, long long sB2, long long sC1, long long sC2)
{
  __shared__ alignas(16) unsigned int As[2][64][16];   // bf16x2, A tile
  __shared__ alignas(16) unsigned int Bt[2][64][16];   // bf16x2, B^T tile: [n][kpair]
  const int z = blockIdx.z;
  const long long zq = z / zmod, zr = z % zmod;
  const float* A = Ab + zq * sA1 + zr * sA2;
  const float* B = Bb + zq * sB1 + zr * sB2;
  float*       C = Cb + zq * sC1 + zr * sC2;

  const int n0 = blockIdx.x * 64;
  const int m0 = blockIdx.y * 64;
  const int t    = threadIdx.x;
  const int lane = t & 31, wave = t >> 5;
  const int wm = wave & 3, wn = wave >> 2;     // 4 m-tiles x 2 n-groups
  const int lr = lane & 15, kh = lane >> 4;

  // staging ownership: A -> 8 consecutive bf16 of one row (one ds_store_b128);
  // B -> 4 n-columns of one k-pair (4 ds_store_b32, packed along K).
  const int ar = t >> 2, ac = (t & 3) * 8;     // A row, bf16 col
  const int kp = t >> 4, bn = (t & 15) * 4;    // B k-pair, n col
  const int agrow = m0 + ar;
  const float* Arow  = A + (long long)agrow * lda + ac;
  const float* Brow0 = B + (long long)(2 * kp) * ldb + n0 + bn;
  const float* Brow1 = Brow0 + ldb;

  float4 a0 = make_float4(0.f,0.f,0.f,0.f), a1 = a0, b0r, b1r;
  if (agrow < M) { a0 = ((const float4*)Arow)[0]; a1 = ((const float4*)Arow)[1]; }
  b0r = *(const float4*)Brow0;
  b1r = *(const float4*)Brow1;
  {
    uint4 w = make_uint4(pack2bf(a0.x,a0.y), pack2bf(a0.z,a0.w),
                         pack2bf(a1.x,a1.y), pack2bf(a1.z,a1.w));
    *(uint4*)&As[0][ar][ac >> 1] = w;
    Bt[0][bn+0][kp] = pack2bf(b0r.x, b1r.x);
    Bt[0][bn+1][kp] = pack2bf(b0r.y, b1r.y);
    Bt[0][bn+2][kp] = pack2bf(b0r.z, b1r.z);
    Bt[0][bn+3][kp] = pack2bf(b0r.w, b1r.w);
  }
  __syncthreads();

  v8f acc0 = {}; v8f acc1 = {};
  for (int kk = 0; kk < K; kk += 32) {
    const int  p    = (kk >> 5) & 1;
    const bool more = (kk + 32) < K;
    if (more) {                       // prefetch next tile into registers
      const int k2 = kk + 32;
      a0 = make_float4(0.f,0.f,0.f,0.f); a1 = a0;
      if (agrow < M) {
        a0 = ((const float4*)(Arow + k2))[0];
        a1 = ((const float4*)(Arow + k2))[1];
      }
      b0r = *(const float4*)(Brow0 + (long long)k2 * ldb);
      b1r = *(const float4*)(Brow1 + (long long)k2 * ldb);
    }
    v16bf a  = frag16(&As[p][wm*16+lr][kh*4],    &As[p][wm*16+lr][8+kh*4]);
    v16bf b0 = frag16(&Bt[p][wn*32+lr][kh*4],    &Bt[p][wn*32+lr][8+kh*4]);
    v16bf b1 = frag16(&Bt[p][wn*32+16+lr][kh*4], &Bt[p][wn*32+16+lr][8+kh*4]);
    acc0 = __builtin_amdgcn_wmma_f32_16x16x32_bf16(false, a, false, b0,
                                                   (short)0, acc0, false, false);
    acc1 = __builtin_amdgcn_wmma_f32_16x16x32_bf16(false, a, false, b1,
                                                   (short)0, acc1, false, false);
    if (more) {
      const int q = p ^ 1;
      uint4 w = make_uint4(pack2bf(a0.x,a0.y), pack2bf(a0.z,a0.w),
                           pack2bf(a1.x,a1.y), pack2bf(a1.z,a1.w));
      *(uint4*)&As[q][ar][ac >> 1] = w;
      Bt[q][bn+0][kp] = pack2bf(b0r.x, b1r.x);
      Bt[q][bn+1][kp] = pack2bf(b0r.y, b1r.y);
      Bt[q][bn+2][kp] = pack2bf(b0r.z, b1r.z);
      Bt[q][bn+3][kp] = pack2bf(b0r.w, b1r.w);
    }
    __syncthreads();
  }
  const int rb = kh * 8;
  #pragma unroll
  for (int r = 0; r < 8; ++r) {
    int row = m0 + wm*16 + r + rb;
    if (row < M) {
      C[(long long)row * ldc + n0 + wn*32 + lr]      = acc0[r];
      C[(long long)row * ldc + n0 + wn*32 + 16 + lr] = acc1[r];
    }
  }
}

// =====================================================================
// Content scores: attn[z=(n,h)][q][v] = SCALE * (q_head + u_h) . k_head[v]
// NT GEMM (C = A.B^T): both tiles stage row-major head slices, K = 64.
// =====================================================================
__global__ __launch_bounds__(256) void score_content(
    const float* __restrict__ qflat, const float* __restrict__ kflat,
    const float* __restrict__ ubias, float* __restrict__ attn)
{
  __shared__ alignas(16) unsigned int As[64][16];
  __shared__ alignas(16) unsigned int Bt[64][16];
  const int z = blockIdx.z;
  const int n = z / H_, h = z % H_;
  const int v0 = blockIdx.x * 64;
  const int q0 = blockIdx.y * 64;
  const int t = threadIdx.x;
  const int lane = t & 31, wave = t >> 5;
  const int wm = wave & 3, wn = wave >> 2;
  const int lr = lane & 15, kh = lane >> 4;

  float* Cz = attn + (long long)z * QS_ * VS_;
  const float* Aq = qflat + (long long)n * QS_ * MD_ + h * D_;
  const float* Bk = kflat + (long long)n * VS_ * MD_ + h * D_;
  const float* ub = ubias + h * D_;

  v8f acc0 = {}; v8f acc1 = {};
  const int r = t >> 2, c = (t & 3) * 8;

  for (int kk = 0; kk < D_; kk += 32) {
    { // stage A = q rows + content bias (one ds_store_b128)
      const float4* ap = (const float4*)(Aq + (long long)(q0 + r) * MD_ + kk + c);
      const float4* up = (const float4*)(ub + kk + c);
      float4 f0 = ap[0], f1 = ap[1], u0 = up[0], u1 = up[1];
      uint4 w = make_uint4(pack2bf(f0.x+u0.x, f0.y+u0.y),
                           pack2bf(f0.z+u0.z, f0.w+u0.w),
                           pack2bf(f1.x+u1.x, f1.y+u1.y),
                           pack2bf(f1.z+u1.z, f1.w+u1.w));
      *(uint4*)&As[r][c >> 1] = w;
    }
    { // stage B^T = k rows directly (one ds_store_b128)
      const float4* bp = (const float4*)(Bk + (long long)(v0 + r) * MD_ + kk + c);
      float4 f0 = bp[0], f1 = bp[1];
      uint4 w = make_uint4(pack2bf(f0.x, f0.y), pack2bf(f0.z, f0.w),
                           pack2bf(f1.x, f1.y), pack2bf(f1.z, f1.w));
      *(uint4*)&Bt[r][c >> 1] = w;
    }
    __syncthreads();
    v16bf a  = frag16(&As[wm*16+lr][kh*4],    &As[wm*16+lr][8+kh*4]);
    v16bf b0 = frag16(&Bt[wn*32+lr][kh*4],    &Bt[wn*32+lr][8+kh*4]);
    v16bf b1 = frag16(&Bt[wn*32+16+lr][kh*4], &Bt[wn*32+16+lr][8+kh*4]);
    acc0 = __builtin_amdgcn_wmma_f32_16x16x32_bf16(false, a, false, b0,
                                                   (short)0, acc0, false, false);
    acc1 = __builtin_amdgcn_wmma_f32_16x16x32_bf16(false, a, false, b1,
                                                   (short)0, acc1, false, false);
    __syncthreads();
  }
  const int rb = kh * 8;
  #pragma unroll
  for (int rr = 0; rr < 8; ++rr) {
    long long row = q0 + wm*16 + rr + rb;
    Cz[row * VS_ + v0 + wn*32 + lr]      = SCALE_ * acc0[rr];
    Cz[row * VS_ + v0 + wn*32 + 16 + lr] = SCALE_ * acc1[rr];
  }
}

// =====================================================================
// Position scores: attn[z][q][v] += SCALE * (q_head + vb_h) . RelP[ridx, h]
// ridx = vpos[v] - qpos[q] + LMAX (general positions; RelP rows hit L2).
// =====================================================================
__global__ __launch_bounds__(256) void score_pos(
    const float* __restrict__ qflat, const float* __restrict__ RelP,
    const float* __restrict__ vbias, const int* __restrict__ qpos,
    const int* __restrict__ vpos, float* __restrict__ attn)
{
  __shared__ alignas(16) float Qs[64][D_];
  __shared__ int qp[64], vp[64];
  const int z = blockIdx.z;
  const int n = z / H_, h = z % H_;
  const int v0 = blockIdx.x * 64;
  const int q0 = blockIdx.y * 64;
  const int t = threadIdx.x;

  if (t < 64) { qp[t] = qpos[q0 + t]; vp[t] = vpos[v0 + t]; }
  { // stage (q + pos_bias) tile: 64 rows x 64 dims
    int r = t >> 2, c0 = (t & 3) * 16;
    const float4* qrow = (const float4*)(qflat + (long long)(n*QS_ + q0 + r) * MD_ + h*D_ + c0);
    const float4* bb   = (const float4*)(vbias + h*D_ + c0);
    float4* dst = (float4*)&Qs[r][c0];
    #pragma unroll
    for (int j = 0; j < 4; ++j) {
      float4 a = qrow[j], b = bb[j];
      dst[j] = make_float4(a.x+b.x, a.y+b.y, a.z+b.z, a.w+b.w);
    }
  }
  __syncthreads();

  const int ql = t >> 2, vb0 = (t & 3) * 16;
  float* Crow = attn + (long long)z * QS_ * VS_ + (long long)(q0 + ql) * VS_ + v0;
  const float4* qv = (const float4*)&Qs[ql][0];
  const int myqp = qp[ql];
  for (int j = 0; j < 16; ++j) {
    int vl = vb0 + j;
    int ridx = vp[vl] - myqp + LMAX_;
    ridx = ridx < 0 ? 0 : (ridx > 2*LMAX_ ? 2*LMAX_ : ridx);
    const float4* rp = (const float4*)(RelP + (long long)ridx * MD_ + h*D_);
    float acc = 0.f;
    #pragma unroll
    for (int d4 = 0; d4 < D_/4; ++d4) {
      float4 a = qv[d4], b = rp[d4];
      acc = fmaf(a.x, b.x, fmaf(a.y, b.y, fmaf(a.z, b.z, fmaf(a.w, b.w, acc))));
    }
    Crow[vl] += SCALE_ * acc;
  }
}

// =====================================================================
// In-place masked softmax over v; one wave32 per 512-length row.
// Each lane owns 16 consecutive elements -> 4 x b128 loads/stores.
// =====================================================================
__global__ __launch_bounds__(256) void softmax_rows(
    float* __restrict__ attn, const unsigned char* __restrict__ mask)
{
  const int wave = threadIdx.x >> 5, lane = threadIdx.x & 31;
  const long long row = (long long)blockIdx.x * 8 + wave;        // < N*H*QS
  const int n = (int)(row / (H_ * QS_));
  const int q = (int)(row % QS_);
  float* s = attn + row * VS_ + lane * 16;
  const unsigned char* m = mask + ((long long)n * QS_ + q) * VS_ + lane * 16;
  const float NEG = -__builtin_inff();

  float4 v4[4];
  unsigned int mm[4];
  #pragma unroll
  for (int i = 0; i < 4; ++i) {
    v4[i] = ((const float4*)s)[i];
    mm[i] = ((const unsigned int*)m)[i];
  }
  float vals[16];
  float mx = NEG;
  #pragma unroll
  for (int i = 0; i < 4; ++i) {
    const float* f = (const float*)&v4[i];
    #pragma unroll
    for (int j = 0; j < 4; ++j) {
      float x = ((mm[i] >> (8*j)) & 0xFFu) ? f[j] : NEG;
      vals[i*4 + j] = x;
      mx = fmaxf(mx, x);
    }
  }
  #pragma unroll
  for (int o = 16; o >= 1; o >>= 1) mx = fmaxf(mx, __shfl_xor(mx, o, 32));
  float sum = 0.f;
  #pragma unroll
  for (int i = 0; i < 16; ++i) { vals[i] = __expf(vals[i] - mx); sum += vals[i]; }
  #pragma unroll
  for (int o = 16; o >= 1; o >>= 1) sum += __shfl_xor(sum, o, 32);
  const float inv = 1.f / sum;
  #pragma unroll
  for (int i = 0; i < 4; ++i) {
    ((float4*)s)[i] = make_float4(vals[i*4+0]*inv, vals[i*4+1]*inv,
                                  vals[i*4+2]*inv, vals[i*4+3]*inv);
  }
}

// =====================================================================
// attention_dist = mean over heads of attn (float4 per thread)
// =====================================================================
__global__ __launch_bounds__(256) void head_mean(
    const float* __restrict__ attn, float* __restrict__ dist)
{
  long long base = ((long long)blockIdx.x * 256 + threadIdx.x) * 4;  // < N*QS*VS
  int n = (int)(base / ((long long)QS_ * VS_));
  long long rem = base % ((long long)QS_ * VS_);
  const float* p = attn + (long long)n * H_ * QS_ * VS_ + rem;
  float4 s = make_float4(0.f, 0.f, 0.f, 0.f);
  #pragma unroll
  for (int h = 0; h < H_; ++h) {
    float4 x = *(const float4*)(p + (long long)h * QS_ * VS_);
    s.x += x.x; s.y += x.y; s.z += x.z; s.w += x.w;
  }
  const float inv = 1.0f / H_;
  *(float4*)(dist + base) = make_float4(s.x*inv, s.y*inv, s.z*inv, s.w*inv);
}

// =====================================================================
extern "C" void kernel_launch(void* const* d_in, const int* in_sizes, int n_in,
                              void* d_out, int out_size, void* d_ws, size_t ws_size,
                              hipStream_t stream)
{
  (void)in_sizes; (void)n_in; (void)out_size; (void)ws_size;
  const float* Q     = (const float*)d_in[0];
  const float* Kin   = (const float*)d_in[1];
  const float* V     = (const float*)d_in[2];
  const unsigned char* mask = (const unsigned char*)d_in[3];
  const int*   qpos  = (const int*)d_in[4];
  const int*   vpos  = (const int*)d_in[5];
  const float* Wq    = (const float*)d_in[6];
  const float* Wk    = (const float*)d_in[7];
  const float* Wv    = (const float*)d_in[8];
  const float* Wpos  = (const float*)d_in[9];
  const float* cbias = (const float*)d_in[10];
  const float* pbias = (const float*)d_in[11];
  const float* Wout  = (const float*)d_in[12];
  const float* relt  = (const float*)d_in[13];

  // workspace layout (floats)
  float* ws    = (float*)d_ws;
  float* qflat = ws;                                         // 4096 x 512
  float* kflat = qflat + (long long)NB_*QS_*MD_;             // 4096 x 512
  float* vflat = kflat + (long long)NB_*VS_*MD_;             // 4096 x 512
  float* RelP  = vflat + (long long)NB_*VS_*MD_;             // 2049 x 512
  float* ctx   = RelP  + (long long)RELROWS_*MD_;            // 4096 x 512

  // output layout: attended | attention_dist | attn
  float* attended = (float*)d_out;
  float* dist     = attended + (long long)NB_*QS_*MD_;
  float* attn     = dist     + (long long)NB_*QS_*VS_;

  dim3 blk(256);
  // 1) projections (bf16 WMMA GEMMs)
  gemm_nn<<<dim3(8, 64, 1), blk, 0, stream>>>(Q,   Wq,   qflat, NB_*QS_, MD_, MD_,
      MD_, MD_, MD_, 1, 0,0,0,0,0,0);
  gemm_nn<<<dim3(8, 64, 1), blk, 0, stream>>>(Kin, Wk,   kflat, NB_*VS_, MD_, MD_,
      MD_, MD_, MD_, 1, 0,0,0,0,0,0);
  gemm_nn<<<dim3(8, 64, 1), blk, 0, stream>>>(V,   Wv,   vflat, NB_*VS_, MD_, MD_,
      MD_, MD_, MD_, 1, 0,0,0,0,0,0);
  gemm_nn<<<dim3(8, 33, 1), blk, 0, stream>>>(relt, Wpos, RelP, RELROWS_, MD_, MD_,
      MD_, MD_, MD_, 1, 0,0,0,0,0,0);
  // 2) scores straight into the attn region of d_out
  score_content<<<dim3(8, 8, NB_*H_), blk, 0, stream>>>(qflat, kflat, cbias, attn);
  score_pos    <<<dim3(8, 8, NB_*H_), blk, 0, stream>>>(qflat, RelP, pbias, qpos, vpos, attn);
  // 3) softmax + head-mean
  softmax_rows<<<dim3(NB_*H_*QS_/8), blk, 0, stream>>>(attn, mask);
  head_mean   <<<dim3((NB_*QS_*VS_)/(256*4)), blk, 0, stream>>>(attn, dist);
  // 4) attended = (attn @ v) @ Wout   (batched z = n*H + h, then flat GEMM)
  gemm_nn<<<dim3(1, 8, NB_*H_), blk, 0, stream>>>(attn, vflat, ctx,
      QS_, D_, VS_, VS_, MD_, MD_,
      H_, (long long)H_*QS_*VS_, (long long)QS_*VS_,
      (long long)VS_*MD_, (long long)D_,
      (long long)QS_*MD_, (long long)D_);
  gemm_nn<<<dim3(8, 64, 1), blk, 0, stream>>>(ctx, Wout, attended, NB_*QS_, MD_, MD_,
      MD_, MD_, MD_, 1, 0,0,0,0,0,0);
}
